// MultiHeadSelfAttention_85856396247667
// MI455X (gfx1250) — matrix operand order repaired and checked by reading, not compiled, not run on hardware
//
#include <hip/hip_runtime.h>

// ---------------------------------------------------------------------------
// MI455X (gfx1250) multi-head self-attention, bf16 WMMA pipeline.
// D=1024, H=16, depth=64, B=4, S=2048.
// ---------------------------------------------------------------------------

#define D_MODEL   1024
#define NUM_HEADS 16
#define DEPTH     64
#define BATCH     4
#define SEQ       2048
#define BH        (BATCH * NUM_HEADS)          // 64
#define MROWS     (BATCH * SEQ)                // 8192
#define NEG_INF_SCALE (-1e9f)

typedef __attribute__((ext_vector_type(16))) __bf16 v16bf;
typedef __attribute__((ext_vector_type(8)))  float  v8f;
typedef int v4i __attribute__((vector_size(16)));
typedef __attribute__((address_space(1))) v4i* gv4i_ptr;
typedef __attribute__((address_space(3))) v4i* lv4i_ptr;

#if __has_builtin(__builtin_amdgcn_global_load_async_to_lds_b128) && \
    __has_builtin(__builtin_amdgcn_s_wait_asynccnt)
#define USE_ASYNC_LDS 1
#else
#define USE_ASYNC_LDS 0
#endif

// A-fragment (16x32 bf16): dword j of lane L holds K = (j>>2)*16 + (L>>4)*8 + (j&3)*2, +1 ; row M = L&15
// B-fragment (32x16 bf16): dword j of lane L holds K = (j>>2)*16 + (L>>4)*8 + (j&3)*2, +1 ; col N = L&15
// C/D (16x16 f32): VGPR r of lane L is element [M = r + (L>>4)*8][N = L&15]

__device__ __forceinline__ v8f wmma_bf16(const v16bf& a, const v16bf& b, const v8f& c) {
    return __builtin_amdgcn_wmma_f32_16x16x32_bf16(false, a, false, b, (short)0, c,
                                                   false, false);
}

// ---------------------------------------------------------------------------
// Weight prep: Wt[n][k] = (bf16)W[k][n].  64x64 tiles via LDS, 256 threads.
// Grid = 256 blocks (16 n-tiles x 16 k-tiles).
// ---------------------------------------------------------------------------
__global__ __launch_bounds__(256)
void transpose_convert_kernel(const float* __restrict__ W, __bf16* __restrict__ Wt) {
    __shared__ float tile[64][65];
    const int bn = blockIdx.x & 15;        // n tile
    const int bk = blockIdx.x >> 4;        // k tile
    const int tx = threadIdx.x & 63;
    const int ty = threadIdx.x >> 6;       // 0..3

    #pragma unroll
    for (int r = ty; r < 64; r += 4)
        tile[r][tx] = W[(size_t)(bk * 64 + r) * D_MODEL + bn * 64 + tx];
    __syncthreads();
    #pragma unroll
    for (int r = ty; r < 64; r += 4)
        Wt[(size_t)(bn * 64 + r) * D_MODEL + bk * 64 + tx] = (__bf16)tile[tx][r];
}

// ---------------------------------------------------------------------------
// Stage a 64(n) x 32(k) bf16 tile of Wt into LDS (row stride 40 elems = 80B,
// keeps 16B alignment for b128 while avoiding bank conflicts).
// 256 x 16B chunks, 128 threads -> exactly 2 chunks each (straight-line, no
// divergent loop) via async-load-to-LDS tracked on ASYNCcnt.
// ---------------------------------------------------------------------------
__device__ __forceinline__ void stage_w_tile(const __bf16* __restrict__ Wt,
                                             int n0, int kk, int tid,
                                             __bf16 (*Bt)[40]) {
    const int row0 = tid >> 2;             // 0..31
    const int row1 = row0 + 32;            // 32..63
    const int ch   = (tid & 3) * 8;        // bf16 elems: 0,8,16,24
    const __bf16* g0 = Wt + (size_t)(n0 + row0) * D_MODEL + kk + ch;
    const __bf16* g1 = Wt + (size_t)(n0 + row1) * D_MODEL + kk + ch;
#if USE_ASYNC_LDS
    __builtin_amdgcn_global_load_async_to_lds_b128((gv4i_ptr)g0,
                                                   (lv4i_ptr)&Bt[row0][ch], 0, 0);
    __builtin_amdgcn_global_load_async_to_lds_b128((gv4i_ptr)g1,
                                                   (lv4i_ptr)&Bt[row1][ch], 0, 0);
    __builtin_amdgcn_s_wait_asynccnt(0);
#else
    *(uint4*)&Bt[row0][ch] = *(const uint4*)g0;
    *(uint4*)&Bt[row1][ch] = *(const uint4*)g1;
#endif
    __syncthreads();
}

// Gather one 32x16 B fragment from the staged tile (two contiguous 16B reads).
__device__ __forceinline__ v16bf gather_b(const __bf16 (*Bt)[40], int n, int hf) {
    v16bf bf;
    #pragma unroll
    for (int j = 0; j < 8; ++j) {
        const int k0 = (j >> 2) * 16 + hf * 8 + (j & 3) * 2;
        bf[2 * j]     = Bt[n][k0];
        bf[2 * j + 1] = Bt[n][k0 + 1];
    }
    return bf;
}

// ---------------------------------------------------------------------------
// Projection GEMM: out = X[8192,1024](f32) * Wt^T + bias, stored bf16 with
// head-split layout. mode 0/1: qh/kh row-major [bh][s][64]; mode 2: vh
// transposed [bh][64][s].  Block = 128 threads = 4 waves, tile 64(M) x 64(N).
// ---------------------------------------------------------------------------
__global__ __launch_bounds__(128)
void proj_gemm_kernel(const float* __restrict__ X, const __bf16* __restrict__ Wt,
                      const float* __restrict__ bias, __bf16* __restrict__ out,
                      int mode) {
    __shared__ __bf16 Bt[64][40];

    const int tile = blockIdx.x;          // 2048 tiles = 128 (M) x 16 (N)
    const int n0   = (tile & 15) * 64;
    const int m0   = (tile >> 4) * 64;
    const int tid  = threadIdx.x;
    const int wv   = tid >> 5;
    const int lane = tid & 31;
    const int lN   = lane & 15;
    const int hf   = lane >> 4;

    v8f acc[4] = {};

    for (int kk = 0; kk < D_MODEL; kk += 32) {
        stage_w_tile(Wt, n0, kk, tid, Bt);

        // A fragment: 16 rows of X, convert fp32 -> bf16 (two 32B runs/lane)
        v16bf a;
        const float* arow = X + (size_t)(m0 + wv * 16 + lN) * D_MODEL + kk;
        #pragma unroll
        for (int j = 0; j < 8; ++j) {
            const int base = (j >> 2) * 16 + hf * 8 + (j & 3) * 2;
            a[2 * j]     = (__bf16)arow[base];
            a[2 * j + 1] = (__bf16)arow[base + 1];
        }
        #pragma unroll
        for (int t = 0; t < 4; ++t) {
            const v16bf bf = gather_b(Bt, t * 16 + lN, hf);
            acc[t] = wmma_bf16(a, bf, acc[t]);
        }
        __syncthreads();
    }

    // store with head split
    #pragma unroll
    for (int t = 0; t < 4; ++t) {
        const int ng = n0 + t * 16 + lN;
        const int hh = ng >> 6, dd = ng & 63;
        const float bval = bias[ng];
        #pragma unroll
        for (int r = 0; r < 8; ++r) {
            const int mg = m0 + wv * 16 + hf * 8 + r;
            const int bb = mg >> 11, ss = mg & 2047;
            const float val = acc[t][r] + bval;
            size_t idx;
            if (mode == 2)   // vh transposed: [bh][d][s]
                idx = (((size_t)(bb * NUM_HEADS + hh)) * DEPTH + dd) * SEQ + ss;
            else             // qh / kh: [bh][s][d]
                idx = (((size_t)(bb * NUM_HEADS + hh)) * SEQ + ss) * DEPTH + dd;
            out[idx] = (__bf16)val;
        }
    }
}

// ---------------------------------------------------------------------------
// Logits: attn[bh][q][k] = (qh . kh)/8 + mask*-1e9.  One wave handles a
// 16(q) x 128(k) strip; A fragment reused across 8 WMMAs. 256 thr = 8 waves.
// ---------------------------------------------------------------------------
__global__ __launch_bounds__(256)
void logits_kernel(const __bf16* __restrict__ qh, const __bf16* __restrict__ kh,
                   const float* __restrict__ mask, float* __restrict__ attn) {
    const int gw   = blockIdx.x * 8 + (threadIdx.x >> 5);  // 131072 wave tiles
    const int kt   = gw & 15;
    const int qt   = (gw >> 4) & 127;
    const int bh   = gw >> 11;
    const int lane = threadIdx.x & 31;
    const int lN   = lane & 15;
    const int hf   = lane >> 4;
    const int q0   = qt * 16;
    const int k0c  = kt * 128;
    const int b    = bh >> 4;

    const __bf16* Q = qh + (size_t)bh * SEQ * DEPTH;
    const __bf16* K = kh + (size_t)bh * SEQ * DEPTH;

    v8f acc[8] = {};
    #pragma unroll
    for (int ks = 0; ks < 2; ++ks) {                 // depth 64 = 2 x 32
        v16bf a;
        const __bf16* arow = Q + (size_t)(q0 + lN) * DEPTH + ks * 32;
        #pragma unroll
        for (int j = 0; j < 8; ++j) {
            const int base = (j >> 2) * 16 + hf * 8 + (j & 3) * 2;
            a[2 * j]     = arow[base];
            a[2 * j + 1] = arow[base + 1];
        }
        #pragma unroll
        for (int t = 0; t < 8; ++t) {
            v16bf bf;
            const __bf16* brow = K + (size_t)(k0c + t * 16 + lN) * DEPTH + ks * 32;
            #pragma unroll
            for (int j = 0; j < 8; ++j) {
                const int k0 = (j >> 2) * 16 + hf * 8 + (j & 3) * 2;
                bf[2 * j]     = brow[k0];
                bf[2 * j + 1] = brow[k0 + 1];
            }
            acc[t] = wmma_bf16(a, bf, acc[t]);
        }
    }

    const float scale = 0.125f;                       // 1/sqrt(64)
    #pragma unroll
    for (int t = 0; t < 8; ++t) {
        const int kg   = k0c + t * 16 + lN;
        const float mv = mask[b * SEQ + kg] * NEG_INF_SCALE;
        #pragma unroll
        for (int r = 0; r < 8; ++r) {
            const int qg = q0 + hf * 8 + r;
            attn[((size_t)bh * SEQ + qg) * SEQ + kg] = acc[t][r] * scale + mv;
        }
    }
}

// ---------------------------------------------------------------------------
// Row softmax over 2048 columns, in place. One block (8 waves) per row.
// ---------------------------------------------------------------------------
__global__ __launch_bounds__(256)
void softmax_kernel(float* __restrict__ attn) {
    __shared__ float red[8];
    float* p = attn + (size_t)blockIdx.x * SEQ;
    const int tid = threadIdx.x, lane = tid & 31, wv = tid >> 5;

    float m = -3.4e38f;
    for (int i = tid; i < SEQ; i += 256) m = fmaxf(m, p[i]);
    #pragma unroll
    for (int off = 16; off > 0; off >>= 1) m = fmaxf(m, __shfl_xor(m, off, 32));
    if (lane == 0) red[wv] = m;
    __syncthreads();
    float bm = red[0];
    #pragma unroll
    for (int w = 1; w < 8; ++w) bm = fmaxf(bm, red[w]);
    __syncthreads();

    float s = 0.f;
    for (int i = tid; i < SEQ; i += 256) {
        const float e = __expf(p[i] - bm);
        p[i] = e;
        s += e;
    }
    #pragma unroll
    for (int off = 16; off > 0; off >>= 1) s += __shfl_xor(s, off, 32);
    if (lane == 0) red[wv] = s;
    __syncthreads();
    float tot = 0.f;
    #pragma unroll
    for (int w = 0; w < 8; ++w) tot += red[w];
    const float inv = 1.0f / tot;
    for (int i = tid; i < SEQ; i += 256) p[i] *= inv;
}

// ---------------------------------------------------------------------------
// ctx[bh][q][d] = attn[bh][q][:] (fp32 -> bf16) x vhT[bh][d][:].
// One wave: 16 q rows x full depth 64 (4 accumulators), K loop = 2048.
// Store into merged-head bf16 layout [B*S, 1024] for the output projection.
// ---------------------------------------------------------------------------
__global__ __launch_bounds__(128)
void ctx_kernel(const float* __restrict__ attn, const __bf16* __restrict__ vht,
                __bf16* __restrict__ ctx) {
    const int gw   = blockIdx.x * 4 + (threadIdx.x >> 5);  // 8192 wave tiles
    const int qt   = gw & 127;
    const int bh   = gw >> 7;
    const int lane = threadIdx.x & 31;
    const int lN   = lane & 15;
    const int hf   = lane >> 4;

    const float*  A  = attn + ((size_t)bh * SEQ + qt * 16 + lN) * SEQ;
    const __bf16* Vt = vht + (size_t)bh * DEPTH * SEQ;

    v8f acc[4] = {};
    for (int ks = 0; ks < SEQ / 32; ++ks) {
        v16bf a;
        const float* ar = A + ks * 32;
        #pragma unroll
        for (int j = 0; j < 8; ++j) {
            const int base = (j >> 2) * 16 + hf * 8 + (j & 3) * 2;
            a[2 * j]     = (__bf16)ar[base];
            a[2 * j + 1] = (__bf16)ar[base + 1];
        }
        #pragma unroll
        for (int t = 0; t < 4; ++t) {
            v16bf bf;
            const __bf16* br = Vt + (size_t)(t * 16 + lN) * SEQ + ks * 32;
            #pragma unroll
            for (int j = 0; j < 8; ++j) {
                const int k0 = (j >> 2) * 16 + hf * 8 + (j & 3) * 2;
                bf[2 * j]     = br[k0];
                bf[2 * j + 1] = br[k0 + 1];
            }
            acc[t] = wmma_bf16(a, bf, acc[t]);
        }
    }

    const int b = bh >> 4, h = bh & 15;
    #pragma unroll
    for (int t = 0; t < 4; ++t) {
        const int d = t * 16 + lN;
        #pragma unroll
        for (int r = 0; r < 8; ++r) {
            const int s = qt * 16 + hf * 8 + r;
            ctx[((size_t)(b * SEQ + s)) * D_MODEL + h * DEPTH + d] = (__bf16)acc[t][r];
        }
    }
}

// ---------------------------------------------------------------------------
// Output projection: out = ctx(bf16)[8192,1024] x woT + bo. fp32 output.
// ---------------------------------------------------------------------------
__global__ __launch_bounds__(128)
void outproj_kernel(const __bf16* __restrict__ ctx, const __bf16* __restrict__ Wt,
                    const float* __restrict__ bias, float* __restrict__ out) {
    __shared__ __bf16 Bt[64][40];

    const int tile = blockIdx.x;
    const int n0   = (tile & 15) * 64;
    const int m0   = (tile >> 4) * 64;
    const int tid  = threadIdx.x;
    const int wv   = tid >> 5;
    const int lane = tid & 31;
    const int lN   = lane & 15;
    const int hf   = lane >> 4;

    v8f acc[4] = {};
    for (int kk = 0; kk < D_MODEL; kk += 32) {
        stage_w_tile(Wt, n0, kk, tid, Bt);

        v16bf a;
        const __bf16* arow = ctx + (size_t)(m0 + wv * 16 + lN) * D_MODEL + kk;
        #pragma unroll
        for (int j = 0; j < 8; ++j) {
            const int base = (j >> 2) * 16 + hf * 8 + (j & 3) * 2;
            a[2 * j]     = arow[base];
            a[2 * j + 1] = arow[base + 1];
        }
        #pragma unroll
        for (int t = 0; t < 4; ++t) {
            const v16bf bf = gather_b(Bt, t * 16 + lN, hf);
            acc[t] = wmma_bf16(a, bf, acc[t]);
        }
        __syncthreads();
    }

    #pragma unroll
    for (int t = 0; t < 4; ++t) {
        const int ng = n0 + t * 16 + lN;
        const float bval = bias[ng];
        #pragma unroll
        for (int r = 0; r < 8; ++r) {
            const int mg = m0 + wv * 16 + hf * 8 + r;
            out[(size_t)mg * D_MODEL + ng] = acc[t][r] + bval;
        }
    }
}

// ---------------------------------------------------------------------------
// Launch. inputs: v,k,q,mask,wq,bq,wk,bk,wv,bv,wo,bo. outputs: out | attn.
// ws: qh(16MB) kh(16MB) vhT(16MB) ctx(16MB) bf16 + 4x transposed bf16 weights.
// ---------------------------------------------------------------------------
extern "C" void kernel_launch(void* const* d_in, const int* in_sizes, int n_in,
                              void* d_out, int out_size, void* d_ws, size_t ws_size,
                              hipStream_t stream) {
    const float* v    = (const float*)d_in[0];
    const float* k    = (const float*)d_in[1];
    const float* q    = (const float*)d_in[2];
    const float* mask = (const float*)d_in[3];
    const float* wq   = (const float*)d_in[4];
    const float* bq   = (const float*)d_in[5];
    const float* wk   = (const float*)d_in[6];
    const float* bk   = (const float*)d_in[7];
    const float* wv   = (const float*)d_in[8];
    const float* bv   = (const float*)d_in[9];
    const float* wo   = (const float*)d_in[10];
    const float* bo   = (const float*)d_in[11];

    const size_t headElems = (size_t)BH * SEQ * DEPTH;        // 8M
    const size_t wElems    = (size_t)D_MODEL * D_MODEL;       // 1M
    __bf16* qh   = (__bf16*)d_ws;
    __bf16* kh   = qh + headElems;
    __bf16* vht  = kh + headElems;
    __bf16* ctxb = vht + headElems;
    __bf16* wqt  = ctxb + headElems;
    __bf16* wkt  = wqt + wElems;
    __bf16* wvt  = wkt + wElems;
    __bf16* wot  = wvt + wElems;

    float* out  = (float*)d_out;
    float* attn = out + (size_t)MROWS * D_MODEL;              // 8,388,608 floats in

    // one-time weight transpose + bf16 convert (L2-resident afterwards)
    transpose_convert_kernel<<<256, 256, 0, stream>>>(wq, wqt);
    transpose_convert_kernel<<<256, 256, 0, stream>>>(wk, wkt);
    transpose_convert_kernel<<<256, 256, 0, stream>>>(wv, wvt);
    transpose_convert_kernel<<<256, 256, 0, stream>>>(wo, wot);

    // Q/K/V projections (bf16 WMMA, fused head split; V stored transposed)
    proj_gemm_kernel<<<2048, 128, 0, stream>>>(q, wqt, bq, qh, 0);
    proj_gemm_kernel<<<2048, 128, 0, stream>>>(k, wkt, bk, kh, 1);
    proj_gemm_kernel<<<2048, 128, 0, stream>>>(v, wvt, bv, vht, 2);

    // scaled + masked logits straight into the attn output region
    logits_kernel<<<16384, 256, 0, stream>>>(qh, kh, mask, attn);

    // in-place row softmax
    softmax_kernel<<<BH * SEQ, 256, 0, stream>>>(attn);

    // context = attn x V
    ctx_kernel<<<2048, 128, 0, stream>>>(attn, vht, ctxb);

    // output projection
    outproj_kernel<<<2048, 128, 0, stream>>>(ctxb, wot, bo, out);
}
